// get_idxs_fps_64037962383752
// MI455X (gfx1250) — compile-verified, matrix operand order checked by reading, not run on hardware
//
#include <hip/hip_runtime.h>

// Problem constants from the reference
#define BB 32
#define NN 4096
#define CC 128
#define SS 2048

#define TPB 512                       // 16 wave32 -> 4 waves per SIMD (latency hiding)
#define NWAVE (TPB / 32)
#define TILES_PER_WAVE (NN / 16 / NWAVE)   // 256 tiles of 16 points / 16 waves = 16

typedef __attribute__((ext_vector_type(16))) __bf16 v16bf;
typedef __attribute__((ext_vector_type(8)))  float  v8f;

#define WMMA_BF16(a, b, c) \
  __builtin_amdgcn_wmma_f32_16x16x32_bf16(false, (a), false, (b), (short)0, (c), false, false)

// ---------------------------------------------------------------------------
// Kernel 1: x[B,C,N] -> x0[B,N,C] (fp32 output), bf16 copy for WMMA, ||p||^2
// Column reads are lane-coalesced (adjacent n -> adjacent addresses).
// ---------------------------------------------------------------------------
__global__ __launch_bounds__(256) void fps_prep_kernel(const float* __restrict__ x,
                                                       float* __restrict__ x0,
                                                       __bf16* __restrict__ pb,
                                                       float* __restrict__ norms) {
  const int gid = blockIdx.x * 256 + threadIdx.x;   // gid = b*NN + n
  const int b = gid >> 12;
  const int n = gid & (NN - 1);
  const float* src = x + (size_t)b * CC * NN + n;
  float*  dst = x0 + (size_t)gid * CC;
  __bf16* pd  = pb + (size_t)gid * CC;
  float acc = 0.0f;
#pragma unroll 4
  for (int c = 0; c < CC; ++c) {
    float v = src[(size_t)c * NN];
    dst[c] = v;
    pd[c]  = (__bf16)v;             // RNE convert, feeds v_wmma_f32_16x16x32_bf16
    acc = fmaf(v, v, acc);
  }
  norms[gid] = acc;
}

// ---------------------------------------------------------------------------
// Kernel 2: one workgroup (16 wave32) per batch runs the 2048 sequential FPS
// steps. Per step the GEMV P[4096x128]*c is done with WMMA:
//   A tile = 16 points x 32 dims (lanes 0-15: row M, K 0-15; lanes 16-31:
//            row M, K 16-31 -> contiguous 32B row loads per lane),
//            double-buffered in VGPRs so tile g+1/g+2 loads fly while tile
//            g/g+1's 4 chained WMMAs execute. An opaque asm dependency on a
//            32-bit offset per step stops LICM from hoisting (and spilling)
//            the step-invariant tile loads out of the 2048-step loop while
//            keeping the pointer provably global (global_load_b128, LOADcnt
//            only — no flat/DScnt interference with the LDS dot stores).
//   B tile = centroid broadcast to all 16 columns (lanes 0-15 hold K 0-15,
//            lanes 16-31 hold K 16-31), loaded once per step from LDS.
//   D      = 16x16 f32; every column identical; lanes with (lane&15)==0 own
//            rows M=VGPRr (+8 for upper half) and spill dots to LDS.
// dist[], norms[], dots[] live in LDS (~48KB of 320KB/WGP); A streams from
// the 192MB L2 (whole bf16 dataset = 32MB) every step.
// ---------------------------------------------------------------------------
__global__ __launch_bounds__(TPB) void fps_iter_kernel(const __bf16* __restrict__ pb,
                                                       const float* __restrict__ norms,
                                                       float* __restrict__ idxf,
                                                       int* __restrict__ idxi) {
  __shared__ float s_dist[NN];                  // running min distance (16KB)
  __shared__ float s_dots[NN];                  // per-step dot products (16KB)
  __shared__ float s_norm[NN];                  // ||p||^2 cached       (16KB)
  __shared__ __align__(32) __bf16 s_cvec[CC];   // current centroid (bf16)
  __shared__ float s_cn;
  __shared__ float s_wval[NWAVE];
  __shared__ int   s_warg[NWAVE];
  __shared__ int   s_cur;

  const int tid  = threadIdx.x;
  const int b    = blockIdx.x;
  const int lane = tid & 31;
  const int wave = tid >> 5;
  const int half = lane >> 4;                   // 0: K 0-15, 1: K 16-31 of each chunk
  const int lrow = lane & 15;                   // matrix row within tile

  const __bf16* pbatch = pb + (size_t)b * NN * CC;
  const float*  nbatch = norms + (size_t)b * NN;

  // This wave's A-tile base: tile g, chunk k lives at apbase[g*CC + 2*k]
  // (v16bf units: one tile = 16 rows * 128 dims = 2048 bf16 = 128 v16bf = CC)
  const v16bf* apbase =
      (const v16bf*)(pbatch + ((size_t)(wave * TILES_PER_WAVE) * 16 + lrow) * CC + half * 16);
  const int p0w = wave * TILES_PER_WAVE * 16;   // first point of this wave

  for (int p = tid; p < NN; p += TPB) {
    s_dist[p] = 1.0e10f;
    s_norm[p] = nbatch[p];
  }
  if (tid == 0) s_cur = 0;                      // deterministic first centroid
  __syncthreads();

  for (int t = 0; t < SS; ++t) {
    const int cur = s_cur;
    if (tid == 0) {
      idxf[(size_t)b * SS + t] = (float)cur;    // reference emits idx BEFORE update
      idxi[(size_t)b * SS + t] = cur;
      s_cn = s_norm[cur];
    }
    if (tid < CC) s_cvec[tid] = pbatch[(size_t)cur * CC + tid];
    __syncthreads();

    // B operand (centroid broadcast), loaded once per step
    const v16bf bv0 = *(const v16bf*)(&s_cvec[half * 16]);
    const v16bf bv1 = *(const v16bf*)(&s_cvec[half * 16 + 32]);
    const v16bf bv2 = *(const v16bf*)(&s_cvec[half * 16 + 64]);
    const v16bf bv3 = *(const v16bf*)(&s_cvec[half * 16 + 96]);

    // Opaque zero offset: forbids LICM/spill of the (step-invariant) tile
    // loads out of the t-loop, but keeps the pointer chain in the global
    // address space so the backend selects global_load_b128 (LOADcnt only).
    int ofs = 0;
    asm volatile("" : "+v"(ofs));
    const v16bf* apb = apbase + ofs;

    // Phase A: dots[p] = <P[p], c>, software-pipelined 2-tile rotation
    v16bf xa0 = apb[0], xa1 = apb[2], xa2 = apb[4], xa3 = apb[6];
#pragma unroll 1
    for (int g = 0; g < TILES_PER_WAVE; g += 2) {
      // prefetch tile g+1 into registers while computing tile g
      const v16bf* ap1 = apb + (size_t)(g + 1) * CC;
      v16bf ya0 = ap1[0], ya1 = ap1[2], ya2 = ap1[4], ya3 = ap1[6];
      if (g + 4 < TILES_PER_WAVE)               // warm near caches ahead
        __builtin_prefetch((const void*)(apb + (size_t)(g + 4) * CC), 0, 3);

      v8f acc = {};
      acc = WMMA_BF16(xa0, bv0, acc);
      acc = WMMA_BF16(xa1, bv1, acc);
      acc = WMMA_BF16(xa2, bv2, acc);
      acc = WMMA_BF16(xa3, bv3, acc);
      if (lrow == 0) {
        const int mb = p0w + g * 16 + half * 8; // D: VGPR r -> M=r / M=r+8
#pragma unroll
        for (int r = 0; r < 8; ++r) s_dots[mb + r] = acc[r];
      }

      // prefetch tile g+2 into registers while computing tile g+1
      if (g + 2 < TILES_PER_WAVE) {
        const v16bf* ap2 = apb + (size_t)(g + 2) * CC;
        xa0 = ap2[0]; xa1 = ap2[2]; xa2 = ap2[4]; xa3 = ap2[6];
      }
      v8f acc2 = {};
      acc2 = WMMA_BF16(ya0, bv0, acc2);
      acc2 = WMMA_BF16(ya1, bv1, acc2);
      acc2 = WMMA_BF16(ya2, bv2, acc2);
      acc2 = WMMA_BF16(ya3, bv3, acc2);
      if (lrow == 0) {
        const int mb = p0w + (g + 1) * 16 + half * 8;
#pragma unroll
        for (int r = 0; r < 8; ++r) s_dots[mb + r] = acc2[r];
      }
    }
    __syncthreads();

    // Phase B: dist update + argmax (first-max tie-break like jnp.argmax)
    const float cn = s_cn;
    float bestv = -3.4e38f;
    int   besta = 0;
    for (int p = tid; p < NN; p += TPB) {
      float d  = s_norm[p] + cn - 2.0f * s_dots[p];
      float dm = fminf(s_dist[p], d);
      s_dist[p] = dm;
      if (dm > bestv) { bestv = dm; besta = p; }   // p ascending per thread
    }
#pragma unroll
    for (int off = 16; off > 0; off >>= 1) {       // wave32 butterfly reduce
      float ov = __shfl_xor(bestv, off, 32);
      int   oa = __shfl_xor(besta, off, 32);
      if (ov > bestv || (ov == bestv && oa < besta)) { bestv = ov; besta = oa; }
    }
    if (lane == 0) { s_wval[wave] = bestv; s_warg[wave] = besta; }
    __syncthreads();
    if (tid == 0) {
      float bv = s_wval[0]; int ba = s_warg[0];
#pragma unroll
      for (int w = 1; w < NWAVE; ++w) {
        if (s_wval[w] > bv || (s_wval[w] == bv && s_warg[w] < ba)) {
          bv = s_wval[w]; ba = s_warg[w];
        }
      }
      s_cur = ba;
    }
    __syncthreads();
  }
}

// ---------------------------------------------------------------------------
// Kernel 3: x1[b,s,:] = x0[b, idx[b,s], :]   (row-coalesced gather)
// ---------------------------------------------------------------------------
__global__ __launch_bounds__(128) void fps_gather_kernel(const float* __restrict__ x0,
                                                         const int* __restrict__ idxi,
                                                         float* __restrict__ x1) {
  const int row = blockIdx.x;                   // row = b*SS + s
  const int b   = row >> 11;
  const int src = idxi[row];
  x1[(size_t)row * CC + threadIdx.x] =
      x0[((size_t)b * NN + src) * CC + threadIdx.x];
}

extern "C" void kernel_launch(void* const* d_in, const int* in_sizes, int n_in,
                              void* d_out, int out_size, void* d_ws, size_t ws_size,
                              hipStream_t stream) {
  (void)in_sizes; (void)n_in; (void)out_size; (void)ws_size;

  const float* x = (const float*)d_in[0];               // [B, C, N] fp32

  // Output tuple concatenated flat: x0 | x1 | idxs (as float)
  float* x0   = (float*)d_out;                          // B*N*C
  float* x1   = x0 + (size_t)BB * NN * CC;              // B*S*C
  float* idxf = x1 + (size_t)BB * SS * CC;              // B*S

  // Workspace: bf16 points (32MB) | norms (512KB) | int indices (256KB)
  __bf16* pb    = (__bf16*)d_ws;
  float*  norms = (float*)((char*)d_ws + (size_t)BB * NN * CC * sizeof(__bf16));
  int*    idxi  = (int*)((char*)norms + (size_t)BB * NN * sizeof(float));

  fps_prep_kernel<<<(BB * NN) / 256, 256, 0, stream>>>(x, x0, pb, norms);
  fps_iter_kernel<<<BB, TPB, 0, stream>>>(pb, norms, idxf, idxi);
  fps_gather_kernel<<<BB * SS, 128, 0, stream>>>(x0, idxi, x1);
}
